// LinearModel_73735998538084
// MI455X (gfx1250) — matrix operand order repaired and checked by reading, not compiled, not run on hardware
//
#include <hip/hip_runtime.h>
#include <hip/hip_bf16.h>

typedef _Float16 v16h __attribute__((ext_vector_type(16)));
typedef _Float16 v8h  __attribute__((ext_vector_type(8)));
typedef float    v8f  __attribute__((ext_vector_type(8)));

#define B_DIM   8192
#define D_IN    4096
#define D_OUT   4096
#define KDIM    4096

#define LDS_ROW_H   40                 // 32 halves + 8 pad (80 B row stride, bank-conflict free)
#define LDS_TILE_H  (128 * LDS_ROW_H)  // halves per 128x32 tile
#define LDS_BUF_H   (2 * LDS_TILE_H)   // A tile + B tile per buffer
#define LDS_BUF_B   (LDS_BUF_H * 2)    // bytes per buffer

// ---------------------------------------------------------------- helpers
__device__ __forceinline__ v16h cat16(v8h lo, v8h hi) {
    v16h r;
#pragma unroll
    for (int i = 0; i < 8; ++i) { r[i] = lo[i]; r[i + 8] = hi[i]; }
    return r;
}

// async global -> LDS copy, 16 bytes per lane; tracked by ASYNCcnt
__device__ __forceinline__ void async_b128(unsigned lds_off, const _Float16* gptr) {
    asm volatile("global_load_async_to_lds_b128 %0, %1, off"
                 :: "v"(lds_off), "v"((unsigned long long)(size_t)gptr)
                 : "memory");
}

// ---------------------------------------------------------------- init
__global__ void init_max_kernel(unsigned int* gmax) {
    *gmax = 0u;   // post-ReLU values >= 0: uint-bit order == float order
}

// ---------------------------------------------------------------- weight quant
__global__ __launch_bounds__(256)
void quant_w_kernel(const float* __restrict__ W, const float* __restrict__ bias,
                    const float* __restrict__ act_sf,
                    _Float16* __restrict__ Wq,
                    float* __restrict__ scale, float* __restrict__ biasterm,
                    float* __restrict__ fc_sf_out) {
    const int row = blockIdx.x;
    const int tid = threadIdx.x;
    const float* wr = W + (size_t)row * D_IN;

    float am = 0.f;
    for (int i = tid; i < D_IN; i += 256) am = fmaxf(am, fabsf(wr[i]));

    __shared__ float red[256];
    red[tid] = am;
    __syncthreads();
#pragma unroll
    for (int s = 128; s > 0; s >>= 1) {
        if (tid < s) red[tid] = fmaxf(red[tid], red[tid + s]);
        __syncthreads();
    }
    const float sf = fmaxf(red[0], 1e-8f) / 127.f;
    const float inv_sf = 1.f / sf;

    _Float16* wq = Wq + (size_t)row * D_IN;
    for (int i = tid; i < D_IN; i += 256) {
        float q = rintf(wr[i] * inv_sf);          // RNE == jnp.round
        q = fminf(fmaxf(q, -128.f), 127.f);
        wq[i] = (_Float16)q;                      // integers exact in f16
    }
    if (tid == 0) {
        fc_sf_out[row] = sf;
        const float asf  = act_sf[0];
        const float bsf  = sf * asf;              // bias_sf
        const float bint = rintf(bias[row] / bsf);
        scale[row]    = sf;                       // act_sf cancels in the fold
        biasterm[row] = bint * bsf;
    }
}

// ---------------------------------------------------------------- x -> f16
__global__ __launch_bounds__(256)
void conv_x_kernel(const float* __restrict__ x, _Float16* __restrict__ xh) {
    const size_t i = ((size_t)blockIdx.x * 256 + threadIdx.x) * 4;
    const float4 v = *(const float4*)(x + i);
    _Float16* p = xh + i;
    p[0] = (_Float16)v.x; p[1] = (_Float16)v.y;
    p[2] = (_Float16)v.z; p[3] = (_Float16)v.w;
}

// ---------------------------------------------------------------- WMMA GEMM
// out[m,n] = relu( (sum_k Xh[m,k]*Wq[n,k]) * scale[n] + biasterm[n] )
// block tile 128x128, 8 waves (wave32): 4 along M x 2 along N, wave tile 32x64.
// Operands staged global->LDS with async B128 copies, double buffered.
__global__ __launch_bounds__(256)
void gemm_wmma_kernel(const _Float16* __restrict__ Xh,   // [B_DIM, KDIM]
                      const _Float16* __restrict__ Wq,   // [D_OUT, KDIM]
                      const float* __restrict__ scale,
                      const float* __restrict__ biasterm,
                      float* __restrict__ out,
                      unsigned int* __restrict__ gmax) {
    __shared__ _Float16 lds[2 * LDS_BUF_H];   // 40 KB

    const int tid   = threadIdx.x;
    const int lane  = tid & 31;
    const int wave  = tid >> 5;
    const int waveM = wave & 3;            // 0..3
    const int waveN = wave >> 2;           // 0..1
    const int blockM = blockIdx.y * 128;
    const int blockN = blockIdx.x * 128;

    const int l15 = lane & 15;
    const int hi  = lane >> 4;

    // ---- staging map: 512 16-byte chunks per tile; thread t -> chunks t, t+256
    const int rowC0 = tid >> 2,         segC0 = tid & 3;
    const int rowC1 = (tid + 256) >> 2, segC1 = (tid + 256) & 3;

    const _Float16* gA0 = Xh + (size_t)(blockM + rowC0) * KDIM + segC0 * 8;
    const _Float16* gA1 = Xh + (size_t)(blockM + rowC1) * KDIM + segC1 * 8;
    const _Float16* gB0 = Wq + (size_t)(blockN + rowC0) * KDIM + segC0 * 8;
    const _Float16* gB1 = Wq + (size_t)(blockN + rowC1) * KDIM + segC1 * 8;

    // generic->LDS offset: low 32 bits of the generic address
    const unsigned lA0 = (unsigned)(size_t)&lds[rowC0 * LDS_ROW_H + segC0 * 8];
    const unsigned lA1 = (unsigned)(size_t)&lds[rowC1 * LDS_ROW_H + segC1 * 8];
    const unsigned lB0 = (unsigned)(size_t)&lds[LDS_TILE_H + rowC0 * LDS_ROW_H + segC0 * 8];
    const unsigned lB1 = (unsigned)(size_t)&lds[LDS_TILE_H + rowC1 * LDS_ROW_H + segC1 * 8];

#define ISSUE_TILES(kk, bb) do {                         \
        const unsigned bo = (unsigned)(bb) * LDS_BUF_B;  \
        async_b128(lA0 + bo, gA0 + (kk));                \
        async_b128(lA1 + bo, gA1 + (kk));                \
        async_b128(lB0 + bo, gB0 + (kk));                \
        async_b128(lB1 + bo, gB1 + (kk));                \
    } while (0)

    // ---- fragment read bases (within a buffer), per WMMA VGPR layouts
    const _Float16* fA[2];
    fA[0] = lds + (waveM * 32 + l15)      * LDS_ROW_H;
    fA[1] = lds + (waveM * 32 + 16 + l15) * LDS_ROW_H;
    const _Float16* fB[4];
#pragma unroll
    for (int tj = 0; tj < 4; ++tj)
        fB[tj] = lds + LDS_TILE_H + (waveN * 64 + tj * 16 + l15) * LDS_ROW_H;

    v8f acc[2][4];
#pragma unroll
    for (int ti = 0; ti < 2; ++ti)
#pragma unroll
        for (int tj = 0; tj < 4; ++tj)
#pragma unroll
            for (int r = 0; r < 8; ++r) acc[ti][tj][r] = 0.f;

    ISSUE_TILES(0, 0);

    for (int k = 0; k < KDIM; k += 32) {
        const int buf = (k >> 5) & 1;
        if (k + 32 < KDIM) {
            ISSUE_TILES(k + 32, buf ^ 1);
            asm volatile("s_wait_asynccnt 0x4" ::: "memory");  // current buffer landed
        } else {
            asm volatile("s_wait_asynccnt 0x0" ::: "memory");
        }
        __syncthreads();                                       // publish to all waves

        const int bofs = buf * LDS_BUF_H;                      // in halves
        v16h afrag[2], bfrag[4];
#pragma unroll
        for (int ti = 0; ti < 2; ++ti) {
            // A layout: 16 contiguous K starting at hi*16
            v8h c0 = *(const v8h*)(fA[ti] + bofs + hi * 16);
            v8h c1 = *(const v8h*)(fA[ti] + bofs + hi * 16 + 8);
            afrag[ti] = cat16(c0, c1);
        }
#pragma unroll
        for (int tj = 0; tj < 4; ++tj) {
            // B layout: 16 contiguous K starting at hi*16
            v8h c0 = *(const v8h*)(fB[tj] + bofs + hi * 16);
            v8h c1 = *(const v8h*)(fB[tj] + bofs + hi * 16 + 8);
            bfrag[tj] = cat16(c0, c1);
        }
#pragma unroll
        for (int ti = 0; ti < 2; ++ti)
#pragma unroll
            for (int tj = 0; tj < 4; ++tj)
                acc[ti][tj] = __builtin_amdgcn_wmma_f32_16x16x32_f16(
                    false, afrag[ti], false, bfrag[tj],
                    (short)0, acc[ti][tj], false, false);

        __syncthreads();                                       // done reading buf
    }
#undef ISSUE_TILES

    // ---- epilogue: C/D layout -> lane l: n = l&15, m = (l>>4)*8 + r
    const int mBase = blockM + waveM * 32;
    const int nBase = blockN + waveN * 64;
    float lmax = 0.f;
#pragma unroll
    for (int tj = 0; tj < 4; ++tj) {
        const int n0 = nBase + tj * 16 + l15;
        const float s  = scale[n0];
        const float bt = biasterm[n0];
#pragma unroll
        for (int ti = 0; ti < 2; ++ti) {
            const int m0 = mBase + ti * 16 + hi * 8;
#pragma unroll
            for (int r = 0; r < 8; ++r) {
                float v = fmaxf(acc[ti][tj][r] * s + bt, 0.f);   // ReLU
                lmax = fmaxf(lmax, v);
                out[(size_t)(m0 + r) * D_OUT + n0] = v;
            }
        }
    }

    __shared__ float red[256];
    red[tid] = lmax;
    __syncthreads();
#pragma unroll
    for (int s = 128; s > 0; s >>= 1) {
        if (tid < s) red[tid] = fmaxf(red[tid], red[tid + s]);
        __syncthreads();
    }
    if (tid == 0)
        atomicMax(gmax, __float_as_uint(red[0]));
}

// ---------------------------------------------------------------- finalize scalar
__global__ void finalize_kernel(const unsigned int* __restrict__ gmax,
                                float* __restrict__ sf_ws,
                                float* __restrict__ asf_out) {
    const float xmax = __uint_as_float(*gmax);      // x_min == 0 after ReLU
    const float sf = fmaxf(xmax, 1e-8f) / 255.f;
    *sf_ws = sf;
    *asf_out = sf;
}

// ---------------------------------------------------------------- requant in place
__global__ __launch_bounds__(256)
void requant_kernel(float* __restrict__ out, const float* __restrict__ sf_ptr) {
    const float sf = *sf_ptr;
    const float inv = 1.f / sf;
    const size_t i = ((size_t)blockIdx.x * 256 + threadIdx.x) * 4;
    float4 v = *(float4*)(out + i);
    v.x = fminf(fmaxf(rintf(v.x * inv), 0.f), 255.f) * sf;
    v.y = fminf(fmaxf(rintf(v.y * inv), 0.f), 255.f) * sf;
    v.z = fminf(fmaxf(rintf(v.z * inv), 0.f), 255.f) * sf;
    v.w = fminf(fmaxf(rintf(v.w * inv), 0.f), 255.f) * sf;
    *(float4*)(out + i) = v;
}

// ---------------------------------------------------------------- launch
extern "C" void kernel_launch(void* const* d_in, const int* in_sizes, int n_in,
                              void* d_out, int out_size, void* d_ws, size_t ws_size,
                              hipStream_t stream) {
    const float* x   = (const float*)d_in[0];
    const float* asf = (const float*)d_in[1];
    const float* W   = (const float*)d_in[2];
    const float* b   = (const float*)d_in[3];

    float* out       = (float*)d_out;                       // [B_DIM*D_OUT] x_out
    float* fc_sf_out = out + (size_t)B_DIM * D_OUT;         // [D_OUT]
    float* asf_out   = fc_sf_out + D_OUT;                   // [1]

    char* ws = (char*)d_ws;
    _Float16* Wq   = (_Float16*)ws;                                   // 32 MB
    _Float16* Xh   = (_Float16*)(ws + (size_t)D_OUT * KDIM * 2);      // 64 MB
    float* scale   = (float*)(ws + (size_t)D_OUT * KDIM * 2
                                 + (size_t)B_DIM * KDIM * 2);
    float* biasterm = scale + D_OUT;
    unsigned int* gmax = (unsigned int*)(biasterm + D_OUT);
    float* sf_ws   = (float*)(gmax + 2);

    init_max_kernel<<<1, 1, 0, stream>>>(gmax);

    quant_w_kernel<<<D_OUT, 256, 0, stream>>>(W, b, asf, Wq, scale, biasterm, fc_sf_out);

    const size_t nx = (size_t)B_DIM * KDIM;
    conv_x_kernel<<<(unsigned)(nx / 4 / 256), 256, 0, stream>>>(x, Xh);

    dim3 grid(D_OUT / 128, B_DIM / 128);   // (32, 64)
    gemm_wmma_kernel<<<grid, 256, 0, stream>>>(Xh, Wq, scale, biasterm, out, gmax);

    finalize_kernel<<<1, 1, 0, stream>>>(gmax, sf_ws, asf_out);

    const size_t no = (size_t)B_DIM * D_OUT;
    requant_kernel<<<(unsigned)(no / 4 / 256), 256, 0, stream>>>(out, sf_ws);
}
